// LocalAttention_86131274154403
// MI455X (gfx1250) — matrix operand order repaired and checked by reading, not compiled, not run on hardware
//
#include <hip/hip_runtime.h>

// ---------- types ----------
typedef __attribute__((ext_vector_type(16))) __bf16          bf16x16;
typedef __attribute__((ext_vector_type(2)))  __bf16          bf16x2;
typedef __attribute__((ext_vector_type(8)))  float           f32x8;
typedef __attribute__((ext_vector_type(8)))  unsigned short  us8;   // 16 bytes

union FragU { bf16x16 v; us8 h[2]; };

// f32 -> bf16 round-to-nearest-even (scalar, used only in prep kernel)
__device__ __forceinline__ unsigned short f2bf(float f) {
  unsigned u = __builtin_bit_cast(unsigned, f);
  u += 0x7FFFu + ((u >> 16) & 1u);
  return (unsigned short)(u >> 16);
}

// two f32 -> packed bf16x2 in ONE VALU op.
// clang-22 does not declare __builtin_amdgcn_cvt_pk_bf16_f32 for gfx1250
// (probe: round 3 fell back to the software path), so emit the CDNA
// instruction directly. Low half <- src0, high half <- src1.
__device__ __forceinline__ unsigned cvt_pk_bf16(float lo, float hi) {
#if __has_builtin(__builtin_amdgcn_cvt_pk_bf16_f32)
  bf16x2 p = __builtin_amdgcn_cvt_pk_bf16_f32(lo, hi);
  return __builtin_bit_cast(unsigned, p);
#else
  unsigned r;
  asm("v_cvt_pk_bf16_f32 %0, %1, %2" : "=v"(r) : "v"(lo), "v"(hi));
  return r;
#endif
}

// A fragment (16x32 bf16, MxK) per ISA table:
//  lanes 0-15: M=lane, VGPR0-3 K=0..7, VGPR4-7 K=16..23; lanes 16-31: K offsets +8
__device__ __forceinline__ bf16x16 loadA(const unsigned short* base, int ld,
                                         int m0, int k0, int lane) {
  int m = m0 + (lane & 15);
  int k = k0 + ((lane >> 4) << 3);
  FragU u;
  u.h[0] = *(const us8*)(base + m * ld + k);        // K = k .. k+7
  u.h[1] = *(const us8*)(base + m * ld + k + 16);   // K = k+16 .. k+23
  return u.v;
}

// B fragment (32x16 bf16, KxN) from an [N][K] (transposed) buffer:
//  lanes 0-15: N=lane, K=0..7 then 16..23; lanes 16-31: same N set, K offsets +8
__device__ __forceinline__ bf16x16 loadBT(const unsigned short* base, int ld,
                                          int n0, int k0, int lane) {
  int n = n0 + (lane & 15);
  int k = k0 + ((lane >> 4) << 3);
  FragU u;
  u.h[0] = *(const us8*)(base + n * ld + k);
  u.h[1] = *(const us8*)(base + n * ld + k + 16);
  return u.v;
}

__device__ __forceinline__ f32x8 wmma_bf16(bf16x16 a, bf16x16 b, f32x8 c) {
  return __builtin_amdgcn_wmma_f32_16x16x32_bf16(false, a, false, b,
                                                 (short)0, c, false, false);
}

#define ZACC {0.f, 0.f, 0.f, 0.f, 0.f, 0.f, 0.f, 0.f}

// D-tile (16x16 f32) stores; layout: VGPR r -> row m0+r(+8 hi half), col n0+(lane&15)
__device__ __forceinline__ void storeD_bf16(unsigned short* dst, int ld, int m0, int n0,
                                            const f32x8& acc, float scl, int lane) {
  int col = n0 + (lane & 15);
  int rb  = m0 + ((lane >> 4) << 3);
#pragma unroll
  for (int r = 0; r < 8; r += 2) {
    unsigned p = cvt_pk_bf16(acc[r] * scl, acc[r + 1] * scl);
    dst[(rb + r) * ld + col]     = (unsigned short)(p & 0xFFFFu);
    dst[(rb + r + 1) * ld + col] = (unsigned short)(p >> 16);
  }
}
// transposed store: consecutive rows contiguous -> packed b32 stores
__device__ __forceinline__ void storeDT_bf16(unsigned short* dst, int ld, int m0, int n0,
                                             const f32x8& acc, int lane) {
  int col = n0 + (lane & 15);
  int rb  = m0 + ((lane >> 4) << 3);
  unsigned* d32 = (unsigned*)(dst + col * ld + rb);   // ld, rb even -> 4B aligned
#pragma unroll
  for (int r = 0; r < 8; r += 2) d32[r >> 1] = cvt_pk_bf16(acc[r], acc[r + 1]);
}
__device__ __forceinline__ void storeD_f32(float* dst, int ld, int m0, int n0,
                                           const f32x8& acc, int lane) {
  int col = n0 + (lane & 15);
  int rb  = m0 + ((lane >> 4) << 3);
#pragma unroll
  for (int r = 0; r < 8; ++r) dst[(rb + r) * ld + col] = acc[r];
}

// ---------- prep: convert + transpose weights to bf16 in workspace ----------
// WqkvT: [1536][256] (rows 0..511 q, 512..1023 k, 1024..1535 v);  WoT: [256][512]
__global__ __launch_bounds__(256)
void la_prep(const float* __restrict__ Wq, const float* __restrict__ Wkv,
             const float* __restrict__ Wo,
             unsigned short* __restrict__ WqkvT, unsigned short* __restrict__ WoT) {
  int idx = blockIdx.x * 256 + threadIdx.x;
  const int NQKV = 1536 * 256;
  if (idx < NQKV) {
    int o = idx / 256, c = idx - o * 256;
    float v = (o < 512) ? Wq[c * 512 + o] : Wkv[c * 1024 + (o - 512)];
    WqkvT[idx] = f2bf(v);
  } else {
    int j = idx - NQKV;
    if (j < 256 * 512) {
      int n = j / 512, k = j - n * 512;
      WoT[j] = f2bf(Wo[k * 256 + n]);
    }
  }
}

// ---------- main: one workgroup (8 wave32) per 7x7 window ----------
#define XLD 264   // bf16 stride: 132 dwords/row -> 4-bank shift, conflict-free b128
#define TLD 72    // 64x64 bf16 tiles
#define SLD 68    // f32 scores
#define OLD 520   // 64x512 bf16 merged-head buffer

__global__ __launch_bounds__(256)
void la_main(const float* __restrict__ fmap,
             const unsigned short* __restrict__ WqkvT,
             const unsigned short* __restrict__ WoT,
             const float* __restrict__ bo,
             float* __restrict__ out) {
  __shared__ __attribute__((aligned(16))) unsigned short Xs[64 * XLD];
  __shared__ __attribute__((aligned(16))) unsigned short Qs[64 * TLD];
  __shared__ __attribute__((aligned(16))) unsigned short Ks[64 * TLD];
  __shared__ __attribute__((aligned(16))) unsigned short Vt[64 * TLD];  // [d][token]
  __shared__ __attribute__((aligned(16))) float          Sc[64 * SLD];
  __shared__ __attribute__((aligned(16))) unsigned short Pb[64 * TLD];
  __shared__ __attribute__((aligned(16))) unsigned short Os[64 * OLD];

  const int tid  = threadIdx.x;
  const int lane = tid & 31;
  const int wv   = tid >> 5;
  const int mt   = wv >> 1;          // this wave's M-tile (0..3)
  const int half = wv & 1;           // N-tile group selector
  const int w    = blockIdx.x;
  const int b    = w >> 6;
  const int wx   = (w >> 3) & 7;
  const int wy   = w & 7;
  const long gbase = (((long)b * 56 + wx * 7) * 56 + wy * 7) * 256;

  // Warm L2 with the shared weight panels (global_prefetch_b8)
  __builtin_prefetch(WqkvT + (tid << 8), 0, 1);
  __builtin_prefetch(WoT + (tid << 8), 0, 1);

  // ---- Phase A: window -> bf16 LDS (64x256, rows 49..63 zeroed) ----
  for (int idx = tid; idx < 49 * 64; idx += 256) {
    int t = idx >> 6, c4 = (idx & 63) << 2;
    int ti = t / 7, tj = t - ti * 7;
    const float4 f = *(const float4*)(fmap + gbase + ((long)ti * 56 + tj) * 256 + c4);
    uint2 p; p.x = cvt_pk_bf16(f.x, f.y); p.y = cvt_pk_bf16(f.z, f.w);
    *(uint2*)(Xs + t * XLD + c4) = p;
  }
  for (int idx = tid; idx < 15 * 64; idx += 256) {
    int t = 49 + (idx >> 6), c4 = (idx & 63) << 2;
    uint2 z = {0u, 0u};
    *(uint2*)(Xs + t * XLD + c4) = z;
  }
  __syncthreads();

  for (int h = 0; h < 8; ++h) {
    // ---- B.1: per-head Q,K,V = X @ W; compile-time sections, A reused x2 ----
    {
      // Q (scale 1/8 folded in)
      f32x8 a0 = ZACC, a1 = ZACC;
      int c0 = h * 64 + (half * 2 + 0) * 16;
      int c1 = h * 64 + (half * 2 + 1) * 16;
#pragma unroll
      for (int kk = 0; kk < 8; ++kk) {
        bf16x16 a  = loadA(Xs, XLD, mt * 16, kk * 32, lane);
        bf16x16 b0 = loadBT(WqkvT, 256, c0, kk * 32, lane);
        bf16x16 b1 = loadBT(WqkvT, 256, c1, kk * 32, lane);
        a0 = wmma_bf16(a, b0, a0);
        a1 = wmma_bf16(a, b1, a1);
      }
      storeD_bf16(Qs, TLD, mt * 16, (half * 2 + 0) * 16, a0, 0.125f, lane);
      storeD_bf16(Qs, TLD, mt * 16, (half * 2 + 1) * 16, a1, 0.125f, lane);
    }
    {
      // K
      f32x8 a0 = ZACC, a1 = ZACC;
      int c0 = 512 + h * 64 + (half * 2 + 0) * 16;
      int c1 = 512 + h * 64 + (half * 2 + 1) * 16;
#pragma unroll
      for (int kk = 0; kk < 8; ++kk) {
        bf16x16 a  = loadA(Xs, XLD, mt * 16, kk * 32, lane);
        bf16x16 b0 = loadBT(WqkvT, 256, c0, kk * 32, lane);
        bf16x16 b1 = loadBT(WqkvT, 256, c1, kk * 32, lane);
        a0 = wmma_bf16(a, b0, a0);
        a1 = wmma_bf16(a, b1, a1);
      }
      storeD_bf16(Ks, TLD, mt * 16, (half * 2 + 0) * 16, a0, 1.0f, lane);
      storeD_bf16(Ks, TLD, mt * 16, (half * 2 + 1) * 16, a1, 1.0f, lane);
    }
    {
      // V (stored transposed [d][token]; packed b32 stores)
      f32x8 a0 = ZACC, a1 = ZACC;
      int c0 = 1024 + h * 64 + (half * 2 + 0) * 16;
      int c1 = 1024 + h * 64 + (half * 2 + 1) * 16;
#pragma unroll
      for (int kk = 0; kk < 8; ++kk) {
        bf16x16 a  = loadA(Xs, XLD, mt * 16, kk * 32, lane);
        bf16x16 b0 = loadBT(WqkvT, 256, c0, kk * 32, lane);
        bf16x16 b1 = loadBT(WqkvT, 256, c1, kk * 32, lane);
        a0 = wmma_bf16(a, b0, a0);
        a1 = wmma_bf16(a, b1, a1);
      }
      storeDT_bf16(Vt, TLD, mt * 16, (half * 2 + 0) * 16, a0, lane);
      storeDT_bf16(Vt, TLD, mt * 16, (half * 2 + 1) * 16, a1, lane);
    }
    __syncthreads();

    // ---- B.2: dots = Q @ K^T (1 M-tile x 2 N-tiles per wave, K=64) ----
    {
      f32x8 a0 = ZACC, a1 = ZACC;
      int n0 = (half * 2 + 0) * 16, n1 = (half * 2 + 1) * 16;
#pragma unroll
      for (int kk = 0; kk < 2; ++kk) {
        bf16x16 a  = loadA(Qs, TLD, mt * 16, kk * 32, lane);
        bf16x16 b0 = loadBT(Ks, TLD, n0, kk * 32, lane);
        bf16x16 b1 = loadBT(Ks, TLD, n1, kk * 32, lane);
        a0 = wmma_bf16(a, b0, a0);
        a1 = wmma_bf16(a, b1, a1);
      }
      storeD_f32(Sc, SLD, mt * 16, n0, a0, lane);
      storeD_f32(Sc, SLD, mt * 16, n1, a1, lane);
    }
    __syncthreads();

    // ---- B.3: masked softmax over 49 valid columns ----
    if (tid < 64) {
      int row = tid;
      unsigned* prow = (unsigned*)(Pb + row * TLD);     // TLD even -> aligned
      if (row < 49) {
        float mx = -1e30f;
        for (int j = 0; j < 49; ++j) mx = fmaxf(mx, Sc[row * SLD + j]);
        float sum = 0.f;
        for (int j = 0; j < 49; ++j) {
          float e = __expf(Sc[row * SLD + j] - mx);
          Sc[row * SLD + j] = e;                        // cache exp
          sum += e;
        }
        float inv = __frcp_rn(sum);
        for (int j = 0; j < 48; j += 2)
          prow[j >> 1] = cvt_pk_bf16(Sc[row * SLD + j] * inv,
                                     Sc[row * SLD + j + 1] * inv);
        prow[24] = cvt_pk_bf16(Sc[row * SLD + 48] * inv, 0.f);  // cols 48,49
        for (int j = 25; j < 32; ++j) prow[j] = 0u;             // cols 50..63
      } else {
        for (int j = 0; j < 32; ++j) prow[j] = 0u;
      }
    }
    __syncthreads();

    // ---- B.4: out_h = P @ V -> merged-head buffer ----
    {
      f32x8 a0 = ZACC, a1 = ZACC;
      int n0 = (half * 2 + 0) * 16, n1 = (half * 2 + 1) * 16;
#pragma unroll
      for (int kk = 0; kk < 2; ++kk) {
        bf16x16 a  = loadA(Pb, TLD, mt * 16, kk * 32, lane);
        bf16x16 b0 = loadBT(Vt, TLD, n0, kk * 32, lane);
        bf16x16 b1 = loadBT(Vt, TLD, n1, kk * 32, lane);
        a0 = wmma_bf16(a, b0, a0);
        a1 = wmma_bf16(a, b1, a1);
      }
      storeD_bf16(Os, OLD, mt * 16, h * 64 + n0, a0, 1.0f, lane);
      storeD_bf16(Os, OLD, mt * 16, h * 64 + n1, a1, 1.0f, lane);
    }
    __syncthreads();
  }

  // ---- Phase C: Y = O @ Wo + bo (1 M-tile x 8 N-tiles per wave, K=512) ----
  {
    f32x8 acc[8] = {ZACC, ZACC, ZACC, ZACC, ZACC, ZACC, ZACC, ZACC};
    const int nb = half * 8;
#pragma unroll 4
    for (int kk = 0; kk < 16; ++kk) {
      bf16x16 a = loadA(Os, OLD, mt * 16, kk * 32, lane);
#pragma unroll
      for (int j = 0; j < 8; ++j) {
        bf16x16 bm = loadBT(WoT, 512, (nb + j) * 16, kk * 32, lane);
        acc[j] = wmma_bf16(a, bm, acc[j]);
      }
    }
    // hoisted epilogue: per-row global offsets + validity, per-j bias
    const int rbase = mt * 16 + ((lane >> 4) << 3);
    long rowOff[8];
    bool valid[8];
#pragma unroll
    for (int r = 0; r < 8; ++r) {
      int m = rbase + r;
      valid[r] = (m < 49);
      int ti = m / 7, tj = m - ti * 7;
      rowOff[r] = gbase + ((long)ti * 56 + tj) * 256;
    }
#pragma unroll
    for (int j = 0; j < 8; ++j) {
      int colL   = (nb + j) * 16 + (lane & 15);
      float bias = bo[colL];
#pragma unroll
      for (int r = 0; r < 8; ++r)
        if (valid[r]) out[rowOff[r] + colL] = acc[j][r] + bias;
    }
  }
}

extern "C" void kernel_launch(void* const* d_in, const int* in_sizes, int n_in,
                              void* d_out, int out_size, void* d_ws, size_t ws_size,
                              hipStream_t stream) {
  const float* fmap = (const float*)d_in[0];
  const float* Wq   = (const float*)d_in[1];
  const float* Wkv  = (const float*)d_in[2];
  const float* Wo   = (const float*)d_in[3];
  const float* bo   = (const float*)d_in[4];
  float* out = (float*)d_out;

  unsigned short* WqkvT = (unsigned short*)d_ws;            // 1536*256 bf16
  unsigned short* WoT   = WqkvT + 1536 * 256;               // 256*512 bf16

  la_prep<<<2048, 256, 0, stream>>>(Wq, Wkv, Wo, WqkvT, WoT);
  la_main<<<2048, 256, 0, stream>>>(fmap, WqkvT, WoT, bo, out); // 32*8*8 windows
}
